// SelfAttentionClifford_51797305590141
// MI455X (gfx1250) — matrix operand order repaired and checked by reading, not compiled, not run on hardware
//
#include <hip/hip_runtime.h>
#include <hip/hip_bf16.h>
#include <math.h>

typedef __attribute__((ext_vector_type(16))) __bf16 v16bf;
typedef __attribute__((ext_vector_type(8)))  float  v8f;

#define NHEAD 8
#define DFEAT 16
#define NTOK  2048
#define NBATCH 4
#define EDIM  128      // DFEAT * 8 blades
#define QT    64       // query rows per workgroup (4 waves x 16)
#define KT    64       // keys per inner iteration (4 wmma column tiles, 2 K-groups)
#define L2E   1.4426950408889634f

union FragU { v16bf v; uint4 q[2]; __bf16 h[16]; };

// ---------------------------------------------------------------------------
// Kernel 1: Q/K/V Clifford projections, fp32 -> bf16
// q pre-scaled by (1/sqrt(128)) * log2(e)  (softmax runs in exp2 domain)
// out layout: [bh][n][e], e = d*8 + c
// ---------------------------------------------------------------------------
__global__ __launch_bounds__(128)
void proj_kernel(const float* __restrict__ x,
                 const float* __restrict__ wq, const float* __restrict__ bq,
                 const float* __restrict__ wk, const float* __restrict__ bk,
                 const float* __restrict__ wv, const float* __restrict__ bv,
                 __bf16* __restrict__ qo, __bf16* __restrict__ ko, __bf16* __restrict__ vo)
{
    __shared__ float xs[16 * 8];
    const int token = blockIdx.x;          // 0..8191
    const int tid   = threadIdx.x;         // 0..127 -> output o = h*16 + d
    xs[tid] = x[(size_t)token * 128 + tid];
    __syncthreads();

    const int b  = token >> 11;            // / 2048
    const int nt = token & 2047;
    const int h  = tid >> 4;
    const int d  = tid & 15;
    const size_t obase = (((size_t)(b * NHEAD + h)) * NTOK + nt) * EDIM + (size_t)d * 8;

    const float* ws[3]  = { wq, wk, wv };
    const float* bs[3]  = { bq, bk, bv };
    __bf16*      os[3]  = { qo, ko, vo };

    for (int p = 0; p < 3; ++p) {
        const float* wrow = ws[p] + (size_t)tid * 64;   // [o][16][4] contiguous
        float acc[8];
        #pragma unroll
        for (int c = 0; c < 8; ++c) acc[c] = 0.f;
        #pragma unroll
        for (int i = 0; i < 16; ++i) {
            const float w0 = wrow[i*4+0], w1 = wrow[i*4+1];
            const float w2 = wrow[i*4+2], w3 = wrow[i*4+3];
            acc[0] += xs[i*8+0] * w0;      // grade map {0,1,1,1,2,2,2,3}
            acc[1] += xs[i*8+1] * w1;
            acc[2] += xs[i*8+2] * w1;
            acc[3] += xs[i*8+3] * w1;
            acc[4] += xs[i*8+4] * w2;
            acc[5] += xs[i*8+5] * w2;
            acc[6] += xs[i*8+6] * w2;
            acc[7] += xs[i*8+7] * w3;
        }
        acc[0] += bs[p][tid];              // bias on scalar blade only
        const float scale = (p == 0) ? 0.08838834764831845f * L2E : 1.f;
        union { uint4 u; __bf16 hh[8]; } pk;
        #pragma unroll
        for (int c = 0; c < 8; ++c) pk.hh[c] = (__bf16)(acc[c] * scale);
        *(uint4*)(os[p] + obase) = pk.u;
    }
}

// ---------------------------------------------------------------------------
// Kernel 2: flash attention with bf16 WMMA (f32 accumulate), exp2-domain softmax
// grid = (32 bh, 32 q-tiles), 128 threads = 4 waves, 16 query rows per wave
// ---------------------------------------------------------------------------
__global__ __launch_bounds__(128)
void attn_kernel(const __bf16* __restrict__ q,
                 const __bf16* __restrict__ k,
                 const __bf16* __restrict__ v,
                 const float*  __restrict__ mask,
                 float* __restrict__ o)
{
    __shared__ __bf16        lds_p[4 * 16 * KT];        // 8 KB: per-wave P (16x64) row-major
    __shared__ unsigned int  lds_vt[EDIM * (KT / 2)];   // 16 KB: [e][keypair] dwords

    const int bh   = blockIdx.x;             // 0..31
    const int b    = bh >> 3;
    const int h    = bh & 7;
    const int tid  = threadIdx.x;
    const int wave = tid >> 5;
    const int lane = tid & 31;
    const int half = lane >> 4;
    const int ln   = lane & 15;

    const int qrow0 = blockIdx.y * QT + wave * 16;
    const int myqr  = qrow0 + ln;            // A-fragment row for this lane

    // --- Q A-fragments: 4 chunks of K=32 along e ---
    // A layout (16-bit 16x32): lane<16 holds K {0..7,16..23}; lane>=16 holds {8..15,24..31}
    const __bf16* qrp = q + ((size_t)bh * NTOK + myqr) * EDIM;
    v16bf afq[4];
    #pragma unroll
    for (int c = 0; c < 4; ++c) {
        FragU f;
        f.q[0] = *(const uint4*)(qrp + c * 32 +      half * 8);
        f.q[1] = *(const uint4*)(qrp + c * 32 + 16 + half * 8);
        afq[c] = f.v;
    }

    // constant all-ones B-fragment (for row sums via WMMA)
    FragU ones;
    #pragma unroll
    for (int i = 0; i < 16; ++i) ones.h[i] = (__bf16)1.f;

    v8f oacc[8];
    #pragma unroll
    for (int c = 0; c < 8; ++c)
        #pragma unroll
        for (int j = 0; j < 8; ++j) oacc[c][j] = 0.f;
    float mrow[8], lrow[8];
    #pragma unroll
    for (int j = 0; j < 8; ++j) { mrow[j] = -INFINITY; lrow[j] = 0.f; }

    const float* mbase = mask + (size_t)b * NTOK * NTOK;

    for (int kb = 0; kb < NTOK / KT; ++kb) {
        const int key0 = kb * KT;

        // ---- S = Q * K^T : four 16x16 column tiles (exp2 domain) ----
        v8f st[4];
        #pragma unroll
        for (int t = 0; t < 4; ++t) {
            v8f s;
            #pragma unroll
            for (int j = 0; j < 8; ++j) s[j] = 0.f;
            const int keyn = key0 + t * 16 + ln;          // B column = lane%16
            const __bf16* krp = k + ((size_t)bh * NTOK + keyn) * EDIM;
            #pragma unroll
            for (int c = 0; c < 4; ++c) {
                // B layout (16-bit 32x16): lane<16 holds K {0..7,16..23}; lane>=16 holds {8..15,24..31}
                FragU f;
                f.q[0] = *(const uint4*)(krp + c * 32 +      half * 8);
                f.q[1] = *(const uint4*)(krp + c * 32 + 16 + half * 8);
                s = __builtin_amdgcn_wmma_f32_16x16x32_bf16(
                        false, afq[c], false, f.v, (short)0, s, false, false);
            }
            // add mask * log2(e)  (C layout: row m = j + 8*half, col = lane%16)
            #pragma unroll
            for (int j = 0; j < 8; ++j) {
                const int qr = qrow0 + j + 8 * half;
                s[j] = fmaf(mbase[(size_t)qr * NTOK + keyn], L2E, s[j]);
            }
            st[t] = s;
        }

        // prefetch next K/V blocks (speculative)
        if (kb + 1 < NTOK / KT) {
            __builtin_prefetch(k + ((size_t)bh * NTOK + key0 + KT + ln) * EDIM, 0, 0);
            __builtin_prefetch(v + ((size_t)bh * NTOK + key0 + KT + ln) * EDIM, 0, 0);
        }

        // ---- online softmax: row max via shuffles, p = exp2(s - m) ----
        float scl[8];
        #pragma unroll
        for (int j = 0; j < 8; ++j) {
            float mx = fmaxf(fmaxf(st[0][j], st[1][j]), fmaxf(st[2][j], st[3][j]));
            mx = fmaxf(mx, __shfl_xor(mx, 1, 16));
            mx = fmaxf(mx, __shfl_xor(mx, 2, 16));
            mx = fmaxf(mx, __shfl_xor(mx, 4, 16));
            mx = fmaxf(mx, __shfl_xor(mx, 8, 16));
            const float mnew = fmaxf(mrow[j], mx);
            const float sc = __builtin_amdgcn_exp2f(mrow[j] - mnew);
            #pragma unroll
            for (int t = 0; t < 4; ++t)
                st[t][j] = __builtin_amdgcn_exp2f(st[t][j] - mnew);
            mrow[j] = mnew;
            scl[j]  = sc;
        }
        #pragma unroll
        for (int c = 0; c < 8; ++c)
            #pragma unroll
            for (int j = 0; j < 8; ++j) oacc[c][j] *= scl[j];
        #pragma unroll
        for (int j = 0; j < 8; ++j) lrow[j] *= scl[j];

        __syncthreads();   // previous iteration's LDS reads complete

        // ---- P (f32 C-layout) -> bf16 row-major [16][64] in this wave's LDS slab ----
        __bf16* pw = lds_p + wave * 16 * KT;
        #pragma unroll
        for (int t = 0; t < 4; ++t)
            #pragma unroll
            for (int j = 0; j < 8; ++j)
                pw[(j + 8 * half) * KT + t * 16 + ln] = (__bf16)st[t][j];

        // ---- cooperative V load, pair-pack transpose into [e][keypair] dwords ----
        #pragma unroll
        for (int rep = 0; rep < 4; ++rep) {
            const int cid = rep * 128 + tid;   // 512 chunks: (keypair, 8-wide e-chunk)
            const int kp  = cid >> 4;          // keypair 0..31
            const int e0  = (cid & 15) * 8;
            union { uint4 u; unsigned int w[4]; } a0, a1;
            const size_t vb = ((size_t)bh * NTOK + key0 + 2 * kp) * EDIM + e0;
            a0.u = *(const uint4*)(v + vb);
            a1.u = *(const uint4*)(v + vb + EDIM);
            #pragma unroll
            for (int jj = 0; jj < 4; ++jj) {
                const unsigned int x0 = a0.w[jj], x1 = a1.w[jj];
                lds_vt[(e0 + 2 * jj    ) * 32 + kp] = (x0 & 0xffffu) | (x1 << 16);
                lds_vt[(e0 + 2 * jj + 1) * 32 + kp] = (x0 >> 16) | (x1 & 0xffff0000u);
            }
        }
        __syncthreads();

        // ---- P A-fragments (two K=32 key groups) ----
        FragU pf[2];
        {
            const __bf16* pr = lds_p + wave * 16 * KT + ln * KT;
            #pragma unroll
            for (int g = 0; g < 2; ++g) {
                pf[g].q[0] = *(const uint4*)(pr + g * 32 + half * 16);
                pf[g].q[1] = *(const uint4*)(pr + g * 32 + half * 16 + 8);
            }
        }

        // ---- row sums of P via WMMA against ones (replaces shuffle reduction) ----
        {
            v8f rs;
            #pragma unroll
            for (int j = 0; j < 8; ++j) rs[j] = 0.f;
            rs = __builtin_amdgcn_wmma_f32_16x16x32_bf16(
                     false, pf[0].v, false, ones.v, (short)0, rs, false, false);
            rs = __builtin_amdgcn_wmma_f32_16x16x32_bf16(
                     false, pf[1].v, false, ones.v, (short)0, rs, false, false);
            #pragma unroll
            for (int j = 0; j < 8; ++j) lrow[j] += rs[j];
        }

        // ---- O += P * V : 2 key groups x 8 e-chunks ----
        #pragma unroll
        for (int g = 0; g < 2; ++g) {
            #pragma unroll
            for (int c = 0; c < 8; ++c) {
                FragU bv;
                const uint4* vp =
                    (const uint4*)&lds_vt[(c * 16 + ln) * 32 + g * 16 + half * 8];
                bv.q[0] = vp[0];
                bv.q[1] = vp[1];
                oacc[c] = __builtin_amdgcn_wmma_f32_16x16x32_bf16(
                              false, pf[g].v, false, bv.v, (short)0, oacc[c], false, false);
            }
        }
    }

    // ---- epilogue: normalize and store f32 as [token][h*128 + e] ----
    #pragma unroll
    for (int j = 0; j < 8; ++j) {
        const float inv = 1.f / lrow[j];
        const int qr = qrow0 + j + 8 * half;
        const size_t tokbase = ((size_t)(b * NTOK + qr)) * 1024 + (size_t)h * EDIM;
        #pragma unroll
        for (int c = 0; c < 8; ++c)
            o[tokbase + c * 16 + ln] = oacc[c][j] * inv;
    }
}

// ---------------------------------------------------------------------------
// Kernel 3: output Clifford projection  [token][128][8] -> [token][16][8]
// ---------------------------------------------------------------------------
__global__ __launch_bounds__(128)
void outproj_kernel(const float* __restrict__ ao,
                    const float* __restrict__ wo,   // [16][128][4]
                    const float* __restrict__ bo,   // [16]
                    float* __restrict__ out)
{
    __shared__ float osd[128 * 8];
    const int token = blockIdx.x;
    const int tid   = threadIdx.x;
    const uint4* src = (const uint4*)(ao + (size_t)token * 1024);
    ((uint4*)osd)[tid]       = src[tid];
    ((uint4*)osd)[tid + 128] = src[tid + 128];
    __syncthreads();

    const int oidx = tid >> 3;
    const int c    = tid & 7;
    const int g    = (c == 0) ? 0 : (c < 4) ? 1 : (c < 7) ? 2 : 3;
    const float* wrow = wo + (size_t)oidx * 512 + g;
    float acc = (c == 0) ? bo[oidx] : 0.f;
    #pragma unroll 8
    for (int i = 0; i < 128; ++i)
        acc += osd[i * 8 + c] * wrow[i * 4];
    out[(size_t)token * 128 + tid] = acc;
}

// ---------------------------------------------------------------------------
extern "C" void kernel_launch(void* const* d_in, const int* in_sizes, int n_in,
                              void* d_out, int out_size, void* d_ws, size_t ws_size,
                              hipStream_t stream)
{
    const float* x    = (const float*)d_in[0];
    const float* mask = (const float*)d_in[1];
    const float* wq = (const float*)d_in[2]; const float* bq = (const float*)d_in[3];
    const float* wk = (const float*)d_in[4]; const float* bk = (const float*)d_in[5];
    const float* wv = (const float*)d_in[6]; const float* bv = (const float*)d_in[7];
    const float* wo = (const float*)d_in[8]; const float* bo = (const float*)d_in[9];

    char* ws = (char*)d_ws;
    const size_t qkv_bytes = (size_t)NBATCH * NHEAD * NTOK * EDIM * sizeof(__bf16); // 16.8 MB
    __bf16* qws = (__bf16*)(ws);
    __bf16* kws = (__bf16*)(ws + qkv_bytes);
    __bf16* vws = (__bf16*)(ws + 2 * qkv_bytes);
    float*  ows = (float*) (ws + 3 * qkv_bytes);   // 8192 * 1024 f32 = 33.6 MB

    proj_kernel<<<dim3(NBATCH * NTOK), dim3(128), 0, stream>>>(
        x, wq, bq, wk, bk, wv, bv, qws, kws, vws);

    attn_kernel<<<dim3(NBATCH * NHEAD, NTOK / QT), dim3(128), 0, stream>>>(
        qws, kws, vws, mask, ows);

    outproj_kernel<<<dim3(NBATCH * NTOK), dim3(128), 0, stream>>>(
        ows, wo, bo, (float*)d_out);
}
